// GenARTTokenizationLayer_87308095193415
// MI455X (gfx1250) — matrix operand order repaired and checked by reading, not compiled, not run on hardware
//
#include <hip/hip_runtime.h>
#include <math.h>

// ---------------- problem constants ----------------
#define B_DIM 8
#define S_DIM 4096
#define C_DIM 768
#define C2_DIM 1536
#define M_TOT (B_DIM * S_DIM)      // 32768 tokens
#define NUM_OFF 6                  // effective conv offsets: -2..+3

// ---------------- WMMA tile config ----------------
#define MT 128       // block M tile (8 waves x 16 rows)
#define NT 64        // block N tile (4 wmma tiles per wave)
#define KT 32        // K slice per iteration (bf16 wmma depth)
#define LDA 40       // LDS row stride in ushorts (80B -> 16B-aligned b128 stores)
#define LDB 40

typedef __attribute__((ext_vector_type(16))) __bf16 v16bf;
typedef __attribute__((ext_vector_type(8)))  float  v8f;
typedef __attribute__((ext_vector_type(4)))  unsigned int v4u;
typedef __attribute__((ext_vector_type(8)))  int v8i_t;
typedef __attribute__((ext_vector_type(4)))  int v4i_t;

// ---------------- helpers ----------------
__device__ __forceinline__ unsigned short f2bf(float x) {
    unsigned int u = __float_as_uint(x);
    unsigned int r = u + 0x7FFFu + ((u >> 16) & 1u);   // round-to-nearest-even
    return (unsigned short)(r >> 16);
}
__device__ __forceinline__ float bf2f(unsigned short h) {
    return __uint_as_float(((unsigned int)h) << 16);
}
__device__ __forceinline__ float gelu_exact(float x) {
    return 0.5f * x * (1.0f + erff(x * 0.70710678118654752f));
}

// CDNA5 async memory->LDS DMA (ASYNCcnt-tracked, no VGPR round trip).
__device__ __forceinline__ void async_copy_b128(unsigned lds_off, const void* g) {
    unsigned long long ga = (unsigned long long)g;
    asm volatile("global_load_async_to_lds_b128 %0, %1, off"
                 :: "v"(lds_off), "v"(ga) : "memory");
}
__device__ __forceinline__ void wait_async() {
    asm volatile("s_wait_asynccnt 0x0" ::: "memory");
}
__device__ __forceinline__ unsigned lds_addr_of(const void* p) {
    return (unsigned)(size_t)p;   // low 32 bits of flat addr == LDS byte offset
}

// ---------------- CDNA5 Tensor Data Mover (TDM) ----------------
// 2D bf16 tile DMA: global [tile_rows][tile_k] (row stride row_stride_elems)
// -> LDS rows padded to LDA=40 ushorts via D# pad fields:
//   row = 32 elem * 2B = 64B = 16 DWORDs -> pad_interval=3 (16 dw)
//   pad = 8 ushorts = 4 DWORDs          -> pad_amount=3   (4 dw)
#if __has_builtin(__builtin_amdgcn_tensor_load_to_lds)
#define HAVE_TDM 1
__device__ __forceinline__ void tdm_load_2d_bf16(
    unsigned lds_byte_addr, const void* gsrc,
    int tile_k, int tile_rows, int row_stride_elems,
    int tensor_d0, int tensor_d1) {
    unsigned long long ga = (unsigned long long)gsrc;
    v4u g0;
    g0[0] = 1u;                                        // count=1, user descriptor
    g0[1] = lds_byte_addr;                             // lds_addr
    g0[2] = (unsigned)(ga & 0xFFFFFFFFull);            // global_addr[31:0]
    g0[3] = (unsigned)((ga >> 32) & 0x01FFFFFFull)     // global_addr[56:32]
            | (2u << 30);                              // type=2 ("image")
    v8i_t g1;
    g1[0] = (1 << 16)                                  // data_size = 2 bytes
          | (1 << 20)                                  // pad_enable
          | (3 << 22)                                  // pad_interval = 16 DWORDs
          | (3 << 25);                                 // pad_amount   = 4 DWORDs
    g1[1] = (tensor_d0 & 0xFFFF) << 16;                // tensor_dim0[15:0]
    g1[2] = ((tensor_d0 >> 16) & 0xFFFF)               // tensor_dim0[31:16]
          | ((tensor_d1 & 0xFFFF) << 16);              // tensor_dim1[15:0]
    g1[3] = ((tensor_d1 >> 16) & 0xFFFF)               // tensor_dim1[31:16]
          | ((tile_k & 0xFFFF) << 16);                 // tile_dim0
    g1[4] = (tile_rows & 0xFFFF);                      // tile_dim1 (tile_dim2=0)
    g1[5] = row_stride_elems;                          // tensor_dim0_stride[31:0]
    g1[6] = 0;                                         // stride hi / dim1_stride lo
    g1[7] = 0;
    v4i_t z4 = {0, 0, 0, 0};
#if defined(__clang_major__) && (__clang_major__ >= 23)
    v8i_t z8 = {0, 0, 0, 0, 0, 0, 0, 0};
    __builtin_amdgcn_tensor_load_to_lds(g0, g1, z4, z4, z8, 0);
#else
    __builtin_amdgcn_tensor_load_to_lds(g0, g1, z4, z4, 0);
#endif
}
__device__ __forceinline__ void wait_tensor() {
    __builtin_amdgcn_s_wait_tensorcnt(0);
}
#else
#define HAVE_TDM 0
#endif

struct Frag {
    union { v16bf v; unsigned int p[8]; };
};

// A fragment (16x32 bf16): lane half 0 -> K {0..7,16..23}; half 1 -> K {8..15,24..31}
__device__ __forceinline__ void load_frag_a(Frag& f, const unsigned short* tile,
                                            int row, int stride, int half) {
    const unsigned short* rp = tile + row * stride;
#pragma unroll
    for (int j = 0; j < 8; ++j) {
        int k = ((j < 4) ? (2 * j) : (16 + 2 * (j - 4))) + (half ? 8 : 0);
        f.p[j] = *(const unsigned int*)(rp + k);
    }
}
// B fragment (32x16 bf16) from N-major LDS tile [N][KT]:
// lane half 0 -> K {0..7,16..23}; half 1 -> K {8..15,24..31}; column = lane&15.
__device__ __forceinline__ void load_frag_b(Frag& f, const unsigned short* tile,
                                            int col, int stride, int half) {
    const unsigned short* rp = tile + col * stride;
#pragma unroll
    for (int j = 0; j < 8; ++j) {
        int k = ((j < 4) ? (2 * j) : (16 + 2 * (j - 4))) + (half ? 8 : 0);
        f.p[j] = *(const unsigned int*)(rp + k);
    }
}

// ======================================================================
// 0) small prep kernels
// ======================================================================
__global__ __launch_bounds__(256) void build_weff_kernel(
    const float* __restrict__ w2k, const float* __restrict__ w3k,
    const float* __restrict__ w4k, const float* __restrict__ w5k,
    const float* __restrict__ w6k, unsigned short* __restrict__ weff) {
    int idx = blockIdx.x * 256 + threadIdx.x;          // over 6*768*768
    if (idx >= NUM_OFF * C_DIM * C_DIM) return;
    int o = idx % C_DIM;
    int rest = idx / C_DIM;
    int i = rest % C_DIM;
    int d = rest / C_DIM;
    int off = d - 2;
    const float* ws[5] = {w2k, w3k, w4k, w5k, w6k};
    const int ksz[5] = {2, 3, 4, 5, 6};
    float s = 0.f;
#pragma unroll
    for (int q = 0; q < 5; ++q) {
        int j = off + (ksz[q] - 1) / 2;                // tap index within kernel q
        if (j >= 0 && j < ksz[q])
            s += ws[q][((size_t)o * C_DIM + i) * ksz[q] + j];
    }
    weff[idx] = f2bf(s);                               // layout [d][in_ch][out_ch]
}

__global__ void f32_to_bf16_kernel(const float* __restrict__ in,
                                   unsigned short* __restrict__ out, long long n) {
    long long i = (long long)blockIdx.x * blockDim.x + threadIdx.x;
    long long stride = (long long)gridDim.x * blockDim.x;
    for (; i < n; i += stride) out[i] = f2bf(in[i]);
}

__global__ void combine_bias_kernel(const float* b0, const float* b1, const float* b2,
                                    const float* b3, const float* b4, float* out) {
    int c = blockIdx.x * 256 + threadIdx.x;
    if (c < C_DIM) out[c] = (b0[c] + b1[c] + b2[c] + b3[c] + b4[c]) * 0.2f;
}

__global__ void zero_f32_kernel(float* p, long long n) {
    long long i = (long long)blockIdx.x * blockDim.x + threadIdx.x;
    long long stride = (long long)gridDim.x * blockDim.x;
    for (; i < n; i += stride) p[i] = 0.f;
}

// ======================================================================
// 1) conv-as-GEMM: ACC[m][o] = sum_{d=0..5} X[m + d - 2][:] @ Weff[d][:][o]
//    (uses async-LDS DMA on the fully-valid fast path)
// ======================================================================
__global__ __launch_bounds__(256) void conv_gemm_kernel(
    const unsigned short* __restrict__ XBF,     // [M_TOT][C] bf16
    const unsigned short* __restrict__ Weff,    // [6][C][C]  bf16
    float* __restrict__ ACC) {                  // [M_TOT][C] f32
    __shared__ __align__(16) unsigned short ldsA[MT * LDA];
    __shared__ __align__(16) unsigned short ldsB[NT * LDB];
    int tid = threadIdx.x;
    int wave = tid >> 5, lane = tid & 31, half = lane >> 4, l16 = lane & 15;
    int mbase = blockIdx.x * MT, nbase = blockIdx.y * NT;

    int ar = tid >> 1;
    int acol = (tid & 1) << 4;                          // 0 or 16 ushorts
    unsigned aldsb = lds_addr_of(ldsA) + (unsigned)(ar * LDA + acol) * 2;
    int bkk = tid >> 3;                                 // 0..31
    int bn0 = (tid & 7) * 8;                            // 0..56

    v8f acc[4] = {};
    for (int d = 0; d < NUM_OFF; ++d) {
        const unsigned short* Bd = Weff + (size_t)d * (C_DIM * C_DIM);
        int off = d - 2;
        int sb = mbase & (S_DIM - 1);                   // block pos within sequence
        bool tile_valid = (sb + off >= 0) && (sb + (MT - 1) + off < S_DIM);
        for (int k0 = 0; k0 < C_DIM; k0 += KT) {
            __syncthreads();
            // ---- stage shifted A tile (128 x 32 bf16) ----
            if (tile_valid) {
                const unsigned short* g =
                    XBF + (size_t)(mbase + ar + off) * C_DIM + k0 + acol;
                async_copy_b128(aldsb, g);
                async_copy_b128(aldsb + 16, g + 8);
            } else {
                int tok = ((mbase + ar) & (S_DIM - 1)) + off;
                bool valid = (tok >= 0) && (tok < S_DIM);
                const unsigned short* g =
                    XBF + (size_t)(mbase + ar + (valid ? off : 0)) * C_DIM + k0 + acol;
                uint4 q0 = *(const uint4*)g;
                uint4 q1 = *(const uint4*)(g + 8);
                q0.x = valid ? q0.x : 0u; q0.y = valid ? q0.y : 0u;
                q0.z = valid ? q0.z : 0u; q0.w = valid ? q0.w : 0u;
                q1.x = valid ? q1.x : 0u; q1.y = valid ? q1.y : 0u;
                q1.z = valid ? q1.z : 0u; q1.w = valid ? q1.w : 0u;
                *(uint4*)(ldsA + ar * LDA + acol) = q0;
                *(uint4*)(ldsA + ar * LDA + acol + 8) = q1;
            }
            // ---- stage B tile transposed into [N][K] ----
            {
                union { uint4 q; unsigned short u[8]; } tb;
                tb.q = *(const uint4*)(Bd + (size_t)(k0 + bkk) * C_DIM + nbase + bn0);
#pragma unroll
                for (int j = 0; j < 8; ++j) ldsB[(bn0 + j) * LDB + bkk] = tb.u[j];
            }
            if (k0 + KT < C_DIM)
                __builtin_prefetch(Bd + (size_t)(k0 + KT + bkk) * C_DIM + nbase + bn0, 0, 0);
            if (tile_valid) wait_async();
            __syncthreads();

            Frag af, bf0, bf1, bf2, bf3;
            load_frag_a(af, ldsA + wave * 16 * LDA, l16, LDA, half);
            load_frag_b(bf0, ldsB + 0 * 16 * LDB, l16, LDB, half);
            load_frag_b(bf1, ldsB + 1 * 16 * LDB, l16, LDB, half);
            load_frag_b(bf2, ldsB + 2 * 16 * LDB, l16, LDB, half);
            load_frag_b(bf3, ldsB + 3 * 16 * LDB, l16, LDB, half);
            acc[0] = __builtin_amdgcn_wmma_f32_16x16x32_bf16(
                false, af.v, false, bf0.v, (short)0, acc[0], false, false);
            acc[1] = __builtin_amdgcn_wmma_f32_16x16x32_bf16(
                false, af.v, false, bf1.v, (short)0, acc[1], false, false);
            acc[2] = __builtin_amdgcn_wmma_f32_16x16x32_bf16(
                false, af.v, false, bf2.v, (short)0, acc[2], false, false);
            acc[3] = __builtin_amdgcn_wmma_f32_16x16x32_bf16(
                false, af.v, false, bf3.v, (short)0, acc[3], false, false);
        }
    }
    int mrow = mbase + wave * 16 + half * 8;
#pragma unroll
    for (int j = 0; j < 4; ++j) {
        int ncol = nbase + j * 16 + l16;
#pragma unroll
        for (int r = 0; r < 8; ++r)
            ACC[(size_t)(mrow + r) * C_DIM + ncol] = acc[j][r];
    }
}

// ======================================================================
// 2) generic bf16 GEMM + exact-GELU epilogue; A tile staged by the TDM
// ======================================================================
__global__ __launch_bounds__(256) void gemm_gelu_kernel(
    const unsigned short* __restrict__ A,     // [M][K] bf16
    const unsigned short* __restrict__ Bmat,  // [K][N] bf16
    const float* __restrict__ bias,           // [N]
    unsigned short* __restrict__ outBF,       // [M][N] bf16
    int M, int N, int K) {
    __shared__ __align__(16) unsigned short ldsA[MT * LDA];
    __shared__ __align__(16) unsigned short ldsB[NT * LDB];
    int tid = threadIdx.x;
    int wave = tid >> 5, lane = tid & 31, half = lane >> 4, l16 = lane & 15;
    int mbase = blockIdx.x * MT, nbase = blockIdx.y * NT;

    int bkk = tid >> 3;
    int bn0 = (tid & 7) * 8;
#if !HAVE_TDM
    int ar = tid >> 1;
    int acol = (tid & 1) << 4;
    unsigned aldsb = lds_addr_of(ldsA) + (unsigned)(ar * LDA + acol) * 2;
#endif

    v8f acc[4] = {};
    for (int k0 = 0; k0 < K; k0 += KT) {
        __syncthreads();
#if HAVE_TDM
        // Tensor Data Mover: one descriptor DMAs the whole 128x32 A tile
        // into LDS with hardware row padding (LDA=40). Issued by wave 0 only.
        if (wave == 0) {
            tdm_load_2d_bf16(lds_addr_of(ldsA),
                             A + (size_t)mbase * K + k0,
                             KT, MT, K, K, M);
        }
#else
        {
            const unsigned short* g = A + (size_t)(mbase + (tid >> 1)) * K + k0 + ((tid & 1) << 4);
            async_copy_b128(aldsb, g);
            async_copy_b128(aldsb + 16, g + 8);
        }
#endif
        {   // register transpose for B tile
            union { uint4 q; unsigned short u[8]; } tb;
            tb.q = *(const uint4*)(Bmat + (size_t)(k0 + bkk) * N + nbase + bn0);
#pragma unroll
            for (int j = 0; j < 8; ++j) ldsB[(bn0 + j) * LDB + bkk] = tb.u[j];
        }
        if (k0 + KT < K)
            __builtin_prefetch(Bmat + (size_t)(k0 + KT + bkk) * N + nbase + bn0, 0, 0);
#if HAVE_TDM
        if (wave == 0) wait_tensor();
#else
        wait_async();
#endif
        __syncthreads();

        Frag af, bf0, bf1, bf2, bf3;
        load_frag_a(af, ldsA + wave * 16 * LDA, l16, LDA, half);
        load_frag_b(bf0, ldsB + 0 * 16 * LDB, l16, LDB, half);
        load_frag_b(bf1, ldsB + 1 * 16 * LDB, l16, LDB, half);
        load_frag_b(bf2, ldsB + 2 * 16 * LDB, l16, LDB, half);
        load_frag_b(bf3, ldsB + 3 * 16 * LDB, l16, LDB, half);
        acc[0] = __builtin_amdgcn_wmma_f32_16x16x32_bf16(
            false, af.v, false, bf0.v, (short)0, acc[0], false, false);
        acc[1] = __builtin_amdgcn_wmma_f32_16x16x32_bf16(
            false, af.v, false, bf1.v, (short)0, acc[1], false, false);
        acc[2] = __builtin_amdgcn_wmma_f32_16x16x32_bf16(
            false, af.v, false, bf2.v, (short)0, acc[2], false, false);
        acc[3] = __builtin_amdgcn_wmma_f32_16x16x32_bf16(
            false, af.v, false, bf3.v, (short)0, acc[3], false, false);
    }
    int mrow = mbase + wave * 16 + half * 8;
#pragma unroll
    for (int j = 0; j < 4; ++j) {
        int ncol = nbase + j * 16 + l16;
        float bv = bias[ncol];
#pragma unroll
        for (int r = 0; r < 8; ++r) {
            float v = gelu_exact(acc[j][r] + bv);
            outBF[(size_t)(mrow + r) * N + ncol] = f2bf(v);
        }
    }
}

// ======================================================================
// 3) residual + special-mask + LayerNorm (one block per token)
// ======================================================================
__global__ __launch_bounds__(256) void ln_kernel(
    const float* __restrict__ hidden, float* __restrict__ accio /* in: ACC, out: CNN */,
    const int* __restrict__ special, const float* __restrict__ cbavg,
    const float* __restrict__ ln_g, const float* __restrict__ ln_b,
    unsigned short* __restrict__ hnbf) {
    int m = blockIdx.x;
    int tid = threadIdx.x;
    bool sp = special[m] != 0;
    __shared__ float red[256];

    float vals[3];
#pragma unroll
    for (int i = 0; i < 3; ++i) {
        int c = tid + i * 256;
        size_t idx = (size_t)m * C_DIM + c;
        float h = hidden[idx];
        vals[i] = sp ? (h + accio[idx] * 0.2f + cbavg[c]) : h;
    }
    float s = vals[0] + vals[1] + vals[2];
    red[tid] = s;
    __syncthreads();
    for (int st = 128; st > 0; st >>= 1) {
        if (tid < st) red[tid] += red[tid + st];
        __syncthreads();
    }
    float mu = red[0] * (1.0f / C_DIM);
    __syncthreads();
    float d0 = vals[0] - mu, d1 = vals[1] - mu, d2 = vals[2] - mu;
    red[tid] = d0 * d0 + d1 * d1 + d2 * d2;
    __syncthreads();
    for (int st = 128; st > 0; st >>= 1) {
        if (tid < st) red[tid] += red[tid + st];
        __syncthreads();
    }
    float inv = rsqrtf(red[0] * (1.0f / C_DIM) + 1e-12f);
#pragma unroll
    for (int i = 0; i < 3; ++i) {
        int c = tid + i * 256;
        size_t idx = (size_t)m * C_DIM + c;
        float hn = (vals[i] - mu) * inv * ln_g[c] + ln_b[c];
        hnbf[idx] = f2bf(hn);
        accio[idx] = vals[i];    // CNN (post-mask, pre-LN) kept for merging
    }
}

// ======================================================================
// 4) score head: scores[m] = dot(h2[m], w3) + b3
// ======================================================================
__global__ __launch_bounds__(256) void scores_kernel(
    const unsigned short* __restrict__ h2, const float* __restrict__ w3,
    const float* __restrict__ b3, float* __restrict__ scores) {
    int m = blockIdx.x;
    int tid = threadIdx.x;
    __shared__ float red[256];
    float s = 0.f;
    for (int c = tid; c < C_DIM; c += 256)
        s += bf2f(h2[(size_t)m * C_DIM + c]) * w3[c];
    red[tid] = s;
    __syncthreads();
    for (int st = 128; st > 0; st >>= 1) {
        if (tid < st) red[tid] += red[tid + st];
        __syncthreads();
    }
    if (tid == 0) scores[m] = red[0] + b3[0];
}

// ======================================================================
// 5) gumbel hard decisions + per-batch inclusive scan -> group ids + weights
// ======================================================================
__global__ __launch_bounds__(1024) void scan_kernel(
    const float* __restrict__ scores, const int* __restrict__ special,
    const float* __restrict__ gumbel_u, int* __restrict__ gids,
    float* __restrict__ wtok) {
    int b = blockIdx.x;
    int tid = threadIdx.x;
    __shared__ int lds[1024];
    const float* sc = scores + (size_t)b * S_DIM;
    const int* sp = special + (size_t)b * S_DIM;
    const float* gu = gumbel_u + (size_t)b * (S_DIM - 1) * 2;

    int gs[4];
    float mdp[4];
    int localsum = 0;
#pragma unroll
    for (int i = 0; i < 4; ++i) {
        int t = tid * 4 + i;
        float m = 0.f;
        if (t >= 1) {                                    // md[t-1]
            int idx = t - 1;
            float ms = sc[idx] + sc[idx + 1];
            float g0 = -logf(-logf(gu[idx * 2 + 0]));
            float g1 = -logf(-logf(gu[idx * 2 + 1]));
            float hard = ((ms + g1) > (-ms + g0)) ? 1.f : 0.f;  // argmax == 1
            m = hard * (1.f - (float)sp[idx + 1]);
        }
        mdp[i] = m;
        int g;
        if (t == 0) g = 1;
        else if (t == S_DIM - 1) g = 0;
        else g = (int)(1.f - m);
        gs[i] = g;
        localsum += g;
    }
    lds[tid] = localsum;
    __syncthreads();
    for (int st = 1; st < 1024; st <<= 1) {              // Hillis-Steele scan
        int v = (tid >= st) ? lds[tid - st] : 0;
        __syncthreads();
        lds[tid] += v;
        __syncthreads();
    }
    int run = lds[tid] - localsum;                       // exclusive prefix
#pragma unroll
    for (int i = 0; i < 4; ++i) {
        int t = tid * 4 + i;
        run += gs[i];
        gids[(size_t)b * S_DIM + t] = run - 1;
        float w = (gs[i] == 1 ? 1.f : 0.f) + mdp[i] * (1.f - (float)sp[t]);
        wtok[(size_t)b * S_DIM + t] = w;
    }
}

// ======================================================================
// 6) segment scatter-add + finalize
// ======================================================================
__global__ __launch_bounds__(256) void scatter_kernel(
    const float* __restrict__ cnn, const int* __restrict__ gids,
    const float* __restrict__ wtok, const int* __restrict__ att,
    const int* __restrict__ spec, float* __restrict__ outH,
    float* __restrict__ counts, float* __restrict__ attsum,
    float* __restrict__ specsum) {
    int m = blockIdx.x;
    int b = m >> 12;                                     // / S_DIM
    int gid = gids[m];
    float wt = wtok[m];
    size_t dstRow = (size_t)(b * S_DIM + gid) * C_DIM;
    size_t srcRow = (size_t)m * C_DIM;
    for (int c = threadIdx.x; c < C_DIM; c += 256)
        atomicAdd(outH + dstRow + c, cnn[srcRow + c] * wt);
    if (threadIdx.x == 0) {
        atomicAdd(counts + b * S_DIM + gid, 1.f);
        atomicAdd(attsum + b * S_DIM + gid, (float)att[m]);
        atomicAdd(specsum + b * S_DIM + gid, (float)spec[m]);
    }
}

__global__ __launch_bounds__(256) void finalize_kernel(
    float* __restrict__ outH, const float* __restrict__ counts,
    const float* __restrict__ attsum, const float* __restrict__ specsum,
    int* __restrict__ attOut, int* __restrict__ specOut) {
    int m = blockIdx.x;
    float cnt = counts[m];
    if (cnt < 1.f) cnt = 1.f;
    float inv = 1.f / cnt;
    size_t row = (size_t)m * C_DIM;
    for (int c = threadIdx.x; c < C_DIM; c += 256) outH[row + c] *= inv;
    if (threadIdx.x == 0) {
        attOut[m] = (attsum[m] > 0.f) ? 1 : 0;
        specOut[m] = (specsum[m] > 0.f) ? 1 : 0;
    }
}

// ======================================================================
// launch
// ======================================================================
extern "C" void kernel_launch(void* const* d_in, const int* in_sizes, int n_in,
                              void* d_out, int out_size, void* d_ws, size_t ws_size,
                              hipStream_t stream) {
    const float* hidden = (const float*)d_in[0];
    const int* att      = (const int*)d_in[1];
    const int* spec     = (const int*)d_in[2];
    const float* gumbel = (const float*)d_in[3];
    const float* cw[5]  = {(const float*)d_in[4], (const float*)d_in[5],
                           (const float*)d_in[6], (const float*)d_in[7],
                           (const float*)d_in[8]};
    const float* cb[5]  = {(const float*)d_in[9], (const float*)d_in[10],
                           (const float*)d_in[11], (const float*)d_in[12],
                           (const float*)d_in[13]};
    const float* ln_g = (const float*)d_in[14];
    const float* ln_b = (const float*)d_in[15];
    const float* w1 = (const float*)d_in[16];
    const float* b1 = (const float*)d_in[17];
    const float* w2 = (const float*)d_in[18];
    const float* b2 = (const float*)d_in[19];
    const float* w3 = (const float*)d_in[20];
    const float* b3 = (const float*)d_in[21];

    // ---- workspace layout (256B aligned) ----
    char* ws = (char*)d_ws;
    size_t off = 0;
    auto alloc = [&](size_t bytes) {
        char* p = ws + off;
        off = (off + bytes + 255) & ~(size_t)255;
        return p;
    };
    unsigned short* XBF  = (unsigned short*)alloc((size_t)M_TOT * C_DIM * 2);
    unsigned short* WEFF = (unsigned short*)alloc((size_t)NUM_OFF * C_DIM * C_DIM * 2);
    unsigned short* W1BF = (unsigned short*)alloc((size_t)C_DIM * C2_DIM * 2);
    unsigned short* W2BF = (unsigned short*)alloc((size_t)C2_DIM * C_DIM * 2);
    float* CBAVG         = (float*)alloc(C_DIM * 4);
    float* ACC           = (float*)alloc((size_t)M_TOT * C_DIM * 4);   // -> CNN in-place
    unsigned short* HNBF = (unsigned short*)alloc((size_t)M_TOT * C_DIM * 2);
    unsigned short* H1BF = (unsigned short*)alloc((size_t)M_TOT * C2_DIM * 2);
    float* SCORES        = (float*)alloc((size_t)M_TOT * 4);
    int* GIDS            = (int*)alloc((size_t)M_TOT * 4);
    float* WTOK          = (float*)alloc((size_t)M_TOT * 4);
    float* COUNTS        = (float*)alloc((size_t)M_TOT * 4);
    float* ATTS          = (float*)alloc((size_t)M_TOT * 4);
    float* SPECS         = (float*)alloc((size_t)M_TOT * 4);
    unsigned short* H2BF = XBF;   // reuse: XBF is dead after conv GEMM

    float* outH = (float*)d_out;
    int* attOut = (int*)(outH + (size_t)M_TOT * C_DIM);
    int* specOut = attOut + M_TOT;

    // ---- 0) weight prep / conversions ----
    build_weff_kernel<<<(NUM_OFF * C_DIM * C_DIM + 255) / 256, 256, 0, stream>>>(
        cw[0], cw[1], cw[2], cw[3], cw[4], WEFF);
    f32_to_bf16_kernel<<<2048, 256, 0, stream>>>(w1, W1BF, (long long)C_DIM * C2_DIM);
    f32_to_bf16_kernel<<<2048, 256, 0, stream>>>(w2, W2BF, (long long)C2_DIM * C_DIM);
    f32_to_bf16_kernel<<<4096, 256, 0, stream>>>(hidden, XBF, (long long)M_TOT * C_DIM);
    combine_bias_kernel<<<3, 256, 0, stream>>>(cb[0], cb[1], cb[2], cb[3], cb[4], CBAVG);

    // ---- 1) conv stack as 6 shifted GEMMs (WMMA bf16, async-LDS staging) ----
    conv_gemm_kernel<<<dim3(M_TOT / MT, C_DIM / NT), 256, 0, stream>>>(XBF, WEFF, ACC);

    // ---- 2) residual + mask + LayerNorm ----
    ln_kernel<<<M_TOT, 256, 0, stream>>>(hidden, ACC, spec, CBAVG, ln_g, ln_b, HNBF);

    // ---- 3) MLP: GEMM+GELU x2 (WMMA bf16, TDM staging) ----
    gemm_gelu_kernel<<<dim3(M_TOT / MT, C2_DIM / NT), 256, 0, stream>>>(
        HNBF, W1BF, b1, H1BF, M_TOT, C2_DIM, C_DIM);
    gemm_gelu_kernel<<<dim3(M_TOT / MT, C_DIM / NT), 256, 0, stream>>>(
        H1BF, W2BF, b2, H2BF, M_TOT, C_DIM, C2_DIM);

    // ---- 4) scores ----
    scores_kernel<<<M_TOT, 256, 0, stream>>>(H2BF, w3, b3, SCORES);

    // ---- 5) gumbel hard decisions + group-id scan ----
    scan_kernel<<<B_DIM, 1024, 0, stream>>>(SCORES, spec, gumbel, GIDS, WTOK);

    // ---- 6) zero accumulators, scatter, finalize ----
    zero_f32_kernel<<<4096, 256, 0, stream>>>(outH, (long long)M_TOT * C_DIM);
    zero_f32_kernel<<<512, 256, 0, stream>>>(COUNTS, (long long)3 * M_TOT); // counts+att+spec contiguous
    scatter_kernel<<<M_TOT, 256, 0, stream>>>(ACC, GIDS, WTOK, att, spec,
                                              outH, COUNTS, ATTS, SPECS);
    finalize_kernel<<<M_TOT, 256, 0, stream>>>(outH, COUNTS, ATTS, SPECS, attOut, specOut);
}
